// CrossAttention_35167192219779
// MI455X (gfx1250) — matrix operand order repaired and checked by reading, not compiled, not run on hardware
//
#include <hip/hip_runtime.h>
#include <hip/hip_bf16.h>

typedef __attribute__((ext_vector_type(16))) __bf16 v16bf;
typedef __attribute__((ext_vector_type(8)))  __bf16 v8bf;
typedef __attribute__((ext_vector_type(8)))  float  v8f;

#define E_DIM 256
#define TILE_M 128          // tokens per workgroup in encode/out kernels
#define NQ_TOT 32768        // B*Q = 8*4096
#define NC_TOT 32768        // B*NC
#define NC_B   4096         // context points per batch
#define K_NN   16

// ---------------------------------------------------------------------------
// Wave-level GEMM tile: D[16 x 256] = A[16 x 256] * B[256 x 256] + bias
// A: row-major bf16, row stride 256 (LDS or global). Bt: bf16 transposed
// weights, Bt[n*256 + k]. Per-nt epilogue callback receives v8f accumulator.
// Fragment layouts follow CDNA5 ISA 7.12.2 (16-bit A 16x32, 16-bit B 32x16,
// 32-bit C/D 16x16).
// ---------------------------------------------------------------------------
template <typename Epi>
__device__ __forceinline__ void gemm16(const __bf16* __restrict__ A,
                                       const __bf16* __restrict__ Bt,
                                       const float*  __restrict__ bias,
                                       int lane, Epi&& epi) {
  const int row  = lane & 15;
  const int half = lane >> 4;          // 0 or 1
  // Preload all A fragments for this wave's 16 rows (K = 0..255).
  v16bf a[8];
#pragma unroll
  for (int kt = 0; kt < 8; ++kt) {
    const v8bf lo = *reinterpret_cast<const v8bf*>(A + row * E_DIM + kt * 32 + half * 8);
    const v8bf hi = *reinterpret_cast<const v8bf*>(A + row * E_DIM + kt * 32 + half * 8 + 16);
#pragma unroll
    for (int i = 0; i < 8; ++i) { a[kt][i] = lo[i]; a[kt][i + 8] = hi[i]; }
  }
  for (int nt = 0; nt < 16; ++nt) {
    const float bval = bias[nt * 16 + row];
    v8f acc = { bval, bval, bval, bval, bval, bval, bval, bval };
#pragma unroll
    for (int kt = 0; kt < 8; ++kt) {
      const v8bf blo = *reinterpret_cast<const v8bf*>(
          Bt + (size_t)(nt * 16 + row) * E_DIM + kt * 32 + half * 8);
      const v8bf bhi = *reinterpret_cast<const v8bf*>(
          Bt + (size_t)(nt * 16 + row) * E_DIM + kt * 32 + half * 8 + 16);
      v16bf bf;
#pragma unroll
      for (int i = 0; i < 8; ++i) { bf[i] = blo[i]; bf[i + 8] = bhi[i]; }
      acc = __builtin_amdgcn_wmma_f32_16x16x32_bf16(false, a[kt], false, bf,
                                                    (short)0, acc, false, false);
    }
    epi(nt, acc);
  }
}

// Branchless tanh-approx GELU: tanh(u) = 1 - 2/(exp(2u)+1) via v_exp_f32.
__device__ __forceinline__ float gelu_tanh(float x) {
  const float u = 0.7978845608028654f * (x + 0.044715f * x * x * x);
  const float t = 1.0f - 2.0f / (__expf(2.0f * u) + 1.0f);
  return 0.5f * x * (1.0f + t);
}

// ---------------------------------------------------------------------------
// K0: fp32 weight [K][N] -> bf16 transposed [N][K]
// ---------------------------------------------------------------------------
__global__ __launch_bounds__(256) void wprep_kernel(const float* __restrict__ src,
                                                    __bf16* __restrict__ dst) {
  const int id = blockIdx.x * 256 + threadIdx.x;   // 65536
  const int k = id >> 8, n = id & 255;
  dst[n * 256 + k] = (__bf16)src[id];
}

// ---------------------------------------------------------------------------
// K1: query path -- posenc (W1 gelu W2) then Wq projection. qh out in fp32.
// ---------------------------------------------------------------------------
__global__ __launch_bounds__(256) void encode_query_kernel(
    const float* __restrict__ qpos, const float* __restrict__ rffB,
    const __bf16* __restrict__ Wt1, const float* __restrict__ b1,
    const __bf16* __restrict__ Wt2, const float* __restrict__ b2,
    const __bf16* __restrict__ Wtq, const float* __restrict__ bq,
    float* __restrict__ qh) {
  __shared__ __bf16 sbuf[TILE_M * E_DIM];          // 64 KB
  const int tid = threadIdx.x;
  const int t0  = blockIdx.x * TILE_M;
  // Step 1: RFF features f = [sin(p@B), cos(p@B)] -> LDS (bf16)
  for (int idx = tid; idx < TILE_M * 128; idx += 256) {
    const int i = idx >> 7, j = idx & 127;
    const float* p = qpos + (size_t)(t0 + i) * 3;
    const float pr = p[0] * rffB[j] + p[1] * rffB[128 + j] + p[2] * rffB[256 + j];
    float s, c;
    __sincosf(pr, &s, &c);
    sbuf[i * E_DIM + j]       = (__bf16)s;
    sbuf[i * E_DIM + 128 + j] = (__bf16)c;
  }
  __syncthreads();
  const int wave = tid >> 5, lane = tid & 31;
  const int row = lane & 15, half = lane >> 4;
  __bf16* Arow = sbuf + wave * 16 * E_DIM;         // wave-private 16 rows

  // GEMM1: gelu(f@W1+b1) -> in place (A frags fully preloaded => safe)
  gemm16(Arow, Wt1, b1, lane, [&](int nt, v8f acc) {
#pragma unroll
    for (int r = 0; r < 8; ++r)
      Arow[(half * 8 + r) * E_DIM + nt * 16 + row] = (__bf16)gelu_tanh(acc[r]);
  });
  // GEMM2: h@W2+b2 -> enc in place
  gemm16(Arow, Wt2, b2, lane, [&](int nt, v8f acc) {
#pragma unroll
    for (int r = 0; r < 8; ++r)
      Arow[(half * 8 + r) * E_DIM + nt * 16 + row] = (__bf16)acc[r];
  });
  // GEMM3: enc@Wq+bq -> qh (fp32, global)
  const size_t gbase = (size_t)(t0 + wave * 16);
  gemm16(Arow, Wtq, bq, lane, [&](int nt, v8f acc) {
#pragma unroll
    for (int r = 0; r < 8; ++r)
      qh[(gbase + half * 8 + r) * E_DIM + nt * 16 + row] = acc[r];
  });
}

// ---------------------------------------------------------------------------
// K2: context path -- posenc then Wk and Wv projections (bf16 out).
// ---------------------------------------------------------------------------
__global__ __launch_bounds__(256) void encode_context_kernel(
    const float* __restrict__ cpos, const float* __restrict__ rffB,
    const __bf16* __restrict__ Wt1, const float* __restrict__ b1,
    const __bf16* __restrict__ Wt2, const float* __restrict__ b2,
    const __bf16* __restrict__ Wtk, const float* __restrict__ bk,
    const __bf16* __restrict__ Wtv, const float* __restrict__ bv,
    __bf16* __restrict__ ck, __bf16* __restrict__ cv) {
  __shared__ __bf16 sbuf[TILE_M * E_DIM];
  const int tid = threadIdx.x;
  const int t0  = blockIdx.x * TILE_M;
  for (int idx = tid; idx < TILE_M * 128; idx += 256) {
    const int i = idx >> 7, j = idx & 127;
    const float* p = cpos + (size_t)(t0 + i) * 3;
    const float pr = p[0] * rffB[j] + p[1] * rffB[128 + j] + p[2] * rffB[256 + j];
    float s, c;
    __sincosf(pr, &s, &c);
    sbuf[i * E_DIM + j]       = (__bf16)s;
    sbuf[i * E_DIM + 128 + j] = (__bf16)c;
  }
  __syncthreads();
  const int wave = tid >> 5, lane = tid & 31;
  const int row = lane & 15, half = lane >> 4;
  __bf16* Arow = sbuf + wave * 16 * E_DIM;

  gemm16(Arow, Wt1, b1, lane, [&](int nt, v8f acc) {
#pragma unroll
    for (int r = 0; r < 8; ++r)
      Arow[(half * 8 + r) * E_DIM + nt * 16 + row] = (__bf16)gelu_tanh(acc[r]);
  });
  gemm16(Arow, Wt2, b2, lane, [&](int nt, v8f acc) {
#pragma unroll
    for (int r = 0; r < 8; ++r)
      Arow[(half * 8 + r) * E_DIM + nt * 16 + row] = (__bf16)acc[r];
  });
  const size_t gbase = (size_t)(t0 + wave * 16);
  gemm16(Arow, Wtk, bk, lane, [&](int nt, v8f acc) {
#pragma unroll
    for (int r = 0; r < 8; ++r)
      ck[(gbase + half * 8 + r) * E_DIM + nt * 16 + row] = (__bf16)acc[r];
  });
  gemm16(Arow, Wtv, bv, lane, [&](int nt, v8f acc) {
#pragma unroll
    for (int r = 0; r < 8; ++r)
      cv[(gbase + half * 8 + r) * E_DIM + nt * 16 + row] = (__bf16)acc[r];
  });
}

// ---------------------------------------------------------------------------
// K3: top-16 nearest context points per query. One block = 256 queries of a
// single batch; batch's context positions staged in LDS (48 KB), scanned with
// broadcast (conflict-free) reads. Stable register bubble-insert matches
// top_k tie ordering.
// ---------------------------------------------------------------------------
__global__ __launch_bounds__(256) void knn_kernel(const float* __restrict__ qpos,
                                                  const float* __restrict__ cpos,
                                                  int* __restrict__ knn) {
  __shared__ float spos[NC_B * 3];                 // 48 KB
  const int g = blockIdx.x * 256 + threadIdx.x;    // 0..32767
  const int b = g >> 12;                           // /4096 (block is batch-pure)
  const float* cp = cpos + (size_t)b * NC_B * 3;
  for (int idx = threadIdx.x; idx < NC_B * 3; idx += 256) spos[idx] = cp[idx];
  __syncthreads();

  const float px = qpos[g * 3 + 0], py = qpos[g * 3 + 1], pz = qpos[g * 3 + 2];
  float best[K_NN];
  int   bidx[K_NN];
#pragma unroll
  for (int j = 0; j < K_NN; ++j) { best[j] = 3.4e38f; bidx[j] = 0; }
  for (int c = 0; c < NC_B; ++c) {
    const float dx = spos[c * 3 + 0] - px;
    const float dy = spos[c * 3 + 1] - py;
    const float dz = spos[c * 3 + 2] - pz;
    const float d = dx * dx + dy * dy + dz * dz;
    if (d < best[K_NN - 1]) {
      float dd = d; int cc = c;
#pragma unroll
      for (int j = 0; j < K_NN; ++j) {
        if (dd < best[j]) {
          const float tb = best[j]; const int ti = bidx[j];
          best[j] = dd; bidx[j] = cc; dd = tb; cc = ti;
        }
      }
    }
  }
#pragma unroll
  for (int j = 0; j < K_NN; ++j)
    knn[(size_t)g * K_NN + j] = b * NC_B + bidx[j];   // pre-offset for gather
}

// ---------------------------------------------------------------------------
// K4: wave-per-query attention (8 heads, 16 keys, head dim 32).
// ---------------------------------------------------------------------------
__global__ __launch_bounds__(256) void attn_kernel(const float* __restrict__ qh,
                                                   const __bf16* __restrict__ ck,
                                                   const __bf16* __restrict__ cv,
                                                   const int* __restrict__ knn,
                                                   __bf16* __restrict__ att) {
  const int tid = threadIdx.x, wave = tid >> 5, lane = tid & 31;
  const size_t g = (size_t)blockIdx.x * 8 + wave;
  const int kk = lane & 15, part = lane >> 4;
  const int myidx = knn[g * K_NN + kk];
  const float scale = 0.17677669529663687f;        // 1/sqrt(32)
  const float* qrow = qh + g * E_DIM;
#pragma unroll 1
  for (int h = 0; h < 8; ++h) {
    const int base = h * 32 + part * 16;
    const v8bf* k8 = reinterpret_cast<const v8bf*>(ck + (size_t)myidx * E_DIM + base);
    const v8bf kv0 = k8[0], kv1 = k8[1];           // 2 x 16-byte loads
    float partial = 0.f;
#pragma unroll
    for (int d = 0; d < 8; ++d) partial += qrow[base + d] * (float)kv0[d];
#pragma unroll
    for (int d = 0; d < 8; ++d) partial += qrow[base + 8 + d] * (float)kv1[d];
    partial += __shfl_xor(partial, 16, 32);        // full 32-dim dot, dup halves
    const float s = partial * scale;
    float m = s;
#pragma unroll
    for (int off = 8; off >= 1; off >>= 1) m = fmaxf(m, __shfl_xor(m, off, 32));
    const float e = __expf(s - m);
    float sum = e;
#pragma unroll
    for (int off = 8; off >= 1; off >>= 1) sum += __shfl_xor(sum, off, 32);
    const float a = e / sum;                       // attn weight for key kk
    float acc = 0.f;
#pragma unroll
    for (int k2 = 0; k2 < 16; ++k2) {
      const float ak = __shfl(a, k2, 32);
      const int   ik = __shfl(myidx, k2, 32);
      acc += ak * (float)cv[(size_t)ik * E_DIM + h * 32 + lane];
    }
    att[g * E_DIM + h * 32 + lane] = (__bf16)acc;
  }
}

// ---------------------------------------------------------------------------
// K5: out = att @ Wo + bo (fp32 out), A streamed from global bf16.
// ---------------------------------------------------------------------------
__global__ __launch_bounds__(256) void out_proj_kernel(const __bf16* __restrict__ att,
                                                       const __bf16* __restrict__ Wto,
                                                       const float* __restrict__ bo,
                                                       float* __restrict__ out) {
  const int tid = threadIdx.x, wave = tid >> 5, lane = tid & 31;
  const int row = lane & 15, half = lane >> 4;
  const size_t r0 = (size_t)blockIdx.x * TILE_M + wave * 16;
  const __bf16* A = att + r0 * E_DIM;
  gemm16(A, Wto, bo, lane, [&](int nt, v8f acc) {
#pragma unroll
    for (int r = 0; r < 8; ++r)
      out[(r0 + half * 8 + r) * E_DIM + nt * 16 + row] = acc[r];
  });
}

// ---------------------------------------------------------------------------
extern "C" void kernel_launch(void* const* d_in, const int* in_sizes, int n_in,
                              void* d_out, int out_size, void* d_ws, size_t ws_size,
                              hipStream_t stream) {
  (void)in_sizes; (void)n_in; (void)out_size; (void)ws_size;
  const float* query_pos   = (const float*)d_in[1];
  const float* context_pos = (const float*)d_in[3];
  const float* rff_B = (const float*)d_in[4];
  const float* W1 = (const float*)d_in[5];  const float* b1 = (const float*)d_in[6];
  const float* W2 = (const float*)d_in[7];  const float* b2 = (const float*)d_in[8];
  const float* Wq = (const float*)d_in[9];  const float* bq = (const float*)d_in[10];
  const float* Wk = (const float*)d_in[11]; const float* bk = (const float*)d_in[12];
  const float* Wv = (const float*)d_in[13]; const float* bv = (const float*)d_in[14];
  const float* Wo = (const float*)d_in[15]; const float* bo = (const float*)d_in[16];

  char* wsb = (char*)d_ws;
  const size_t WBYTES = (size_t)256 * 256 * 2;
  size_t off = 0;
  __bf16* Wt1 = (__bf16*)(wsb + off); off += WBYTES;
  __bf16* Wt2 = (__bf16*)(wsb + off); off += WBYTES;
  __bf16* Wtq = (__bf16*)(wsb + off); off += WBYTES;
  __bf16* Wtk = (__bf16*)(wsb + off); off += WBYTES;
  __bf16* Wtv = (__bf16*)(wsb + off); off += WBYTES;
  __bf16* Wto = (__bf16*)(wsb + off); off += WBYTES;
  float*  qh  = (float*)(wsb + off);  off += (size_t)NQ_TOT * E_DIM * 4;
  __bf16* ck  = (__bf16*)(wsb + off); off += (size_t)NC_TOT * E_DIM * 2;
  __bf16* cv  = (__bf16*)(wsb + off); off += (size_t)NC_TOT * E_DIM * 2;
  int*    knn = (int*)(wsb + off);    off += (size_t)NQ_TOT * K_NN * 4;
  __bf16* att = (__bf16*)(wsb + off); off += (size_t)NQ_TOT * E_DIM * 2;

  // K0: weight prep (bf16 + transpose)
  wprep_kernel<<<256, 256, 0, stream>>>(W1, Wt1);
  wprep_kernel<<<256, 256, 0, stream>>>(W2, Wt2);
  wprep_kernel<<<256, 256, 0, stream>>>(Wq, Wtq);
  wprep_kernel<<<256, 256, 0, stream>>>(Wk, Wtk);
  wprep_kernel<<<256, 256, 0, stream>>>(Wv, Wtv);
  wprep_kernel<<<256, 256, 0, stream>>>(Wo, Wto);

  // K1/K2: encode + projections (WMMA)
  encode_query_kernel<<<NQ_TOT / TILE_M, 256, 0, stream>>>(
      query_pos, rff_B, Wt1, b1, Wt2, b2, Wtq, bq, qh);
  encode_context_kernel<<<NC_TOT / TILE_M, 256, 0, stream>>>(
      context_pos, rff_B, Wt1, b1, Wt2, b2, Wtk, bk, Wtv, bv, ck, cv);

  // K3: kNN (LDS-staged candidate scan)
  knn_kernel<<<NQ_TOT / 256, 256, 0, stream>>>(query_pos, context_pos, knn);

  // K4: attention
  attn_kernel<<<NQ_TOT / 8, 256, 0, stream>>>(qh, ck, cv, knn, att);

  // K5: output projection (WMMA)
  out_proj_kernel<<<NQ_TOT / TILE_M, 256, 0, stream>>>(att, Wto, bo, (float*)d_out);

  // Second tuple output: query_pos passthrough
  hipMemcpyAsync((float*)d_out + (size_t)NQ_TOT * E_DIM, query_pos,
                 (size_t)NQ_TOT * 3 * sizeof(float), hipMemcpyDeviceToDevice, stream);
}
